// Net_89584427860581
// MI455X (gfx1250) — matrix operand order repaired and checked by reading, not compiled, not run on hardware
//
#include <hip/hip_runtime.h>
#include <cstdint>

#define NNODES 50000
#define NEDGES 800000
#define ETOT   (NNODES + NEDGES)
#define DMODEL 256
#define SLOPE  0.2f

typedef __bf16 bf16;
typedef __attribute__((ext_vector_type(16))) __bf16   v16bf;
typedef __attribute__((ext_vector_type(8)))  float    v8f;
typedef __attribute__((ext_vector_type(4)))  uint32_t u32x4;
typedef __attribute__((ext_vector_type(4)))  int      v4i;

union Frag { v16bf v; u32x4 q[2]; uint32_t u[8]; };

__device__ inline unsigned f2bf_bits(float f) {
  __bf16 b = (__bf16)f;
  return (unsigned)__builtin_bit_cast(unsigned short, b);
}
__device__ inline unsigned ordf(float f) {
  unsigned u = __float_as_uint(f);
  return (u & 0x80000000u) ? ~u : (u | 0x80000000u);
}
__device__ inline float deordf(unsigned u) {
  unsigned b = (u & 0x80000000u) ? (u & 0x7FFFFFFFu) : ~u;
  return __uint_as_float(b);
}

// ---------------- fp32 -> bf16 cast ----------------
__global__ void cast_bf16_k(const float* __restrict__ in, bf16* __restrict__ out, int n) {
  int i = blockIdx.x * blockDim.x + threadIdx.x;
  if (i < n) out[i] = (bf16)in[i];
}

// ---------------- pack W (K x 256 fp32) into WMMA-B fragment layout ----------------
// Bp linear index = ((kt*16 + ntile)*32 + lane)*8 + v
// lane holds col n = ntile*16 + (lane&15); VGPR v holds K = kt*32 + ((lane&16)?8:0) + ((v&4)?16:0) + 2(v&3), +1
__global__ void pack_B_k(const float* __restrict__ W, uint32_t* __restrict__ Bp, int K) {
  int idx = blockIdx.x * blockDim.x + threadIdx.x;
  int total = (K >> 5) * 16 * 32 * 8;
  if (idx >= total) return;
  int v     = idx & 7;
  int lane  = (idx >> 3) & 31;
  int ntile = (idx >> 8) & 15;
  int kt    = idx >> 12;
  int n = ntile * 16 + (lane & 15);
  int k = kt * 32 + ((lane & 16) ? 8 : 0) + ((v & 4) ? 16 : 0) + 2 * (v & 3);
  unsigned lo = f2bf_bits(W[(size_t)k * DMODEL + n]);
  unsigned hi = f2bf_bits(W[(size_t)(k + 1) * DMODEL + n]);
  Bp[idx] = lo | (hi << 16);
}

// ---------------- bf16 WMMA GEMM: C[M x 256] = A[M x K] * B[K x 256] ----------------
// 128 threads = 4 waves; block computes 16 rows x 256 cols; each wave a 16x64 strip.
// B (already in WMMA fragment layout) is staged into LDS once per block with
// GLOBAL_LOAD_ASYNC_TO_LDS_B128 (ASYNCcnt); the k-loop feeds WMMA from LDS while
// the A fragments are double-buffered from global memory.
__global__ __launch_bounds__(128) void gemm_bf16_wmma(
    const bf16* __restrict__ A, const uint32_t* __restrict__ Bp,
    float* __restrict__ C, int K) {
  extern __shared__ uint32_t smem[];  // (K/32)*16*32*8 dwords = K*512 bytes

  const int lane = threadIdx.x & 31;
  const int wave = threadIdx.x >> 5;
  const int tm   = blockIdx.x;
  const int hi   = lane >> 4;
  const size_t rowoff = (size_t)(tm * 16 + (lane & 15)) * K + hi * 8;

  // ---- stage packed B into LDS (one async b128 copy per 16B chunk) ----
  const int nvec = K * 32;  // (K*128 dwords) / 4 per b128
#if __has_builtin(__builtin_amdgcn_global_load_async_to_lds_b128)
  for (int i = threadIdx.x; i < nvec; i += 128) {
    v4i* gp = (v4i*)(Bp + (size_t)i * 4);    // const-strip + reinterpret (generic)
    v4i* lp = (v4i*)(smem + (size_t)i * 4);  // generic
    __builtin_amdgcn_global_load_async_to_lds_b128(
        (__attribute__((address_space(1))) v4i*)gp,
        (__attribute__((address_space(3))) v4i*)lp, 0, 0);
  }
#if __has_builtin(__builtin_amdgcn_s_wait_asynccnt)
  __builtin_amdgcn_s_wait_asynccnt(0);
#else
  asm volatile("s_wait_asynccnt 0x0" ::: "memory");
#endif
#else
  for (int i = threadIdx.x; i < nvec; i += 128)
    *(u32x4*)(smem + (size_t)i * 4) = *(const u32x4*)(Bp + (size_t)i * 4);
#endif
  __syncthreads();

  const uint32_t* bbase = smem + ((size_t)(wave * 4) * 32 + lane) * 8;

  v8f zero = {0.f, 0.f, 0.f, 0.f, 0.f, 0.f, 0.f, 0.f};
  v8f acc[4];
#pragma unroll
  for (int j = 0; j < 4; ++j) acc[j] = zero;

  Frag a0, a1;

  auto loadA = [&](Frag& a, int kt) {
    const bf16* ap = A + rowoff + kt * 32;
    a.q[0] = *(const u32x4*)(ap);        // VGPR0-3: K = base+0..7
    a.q[1] = *(const u32x4*)(ap + 16);   // VGPR4-7: K = base+16..23
  };
  auto mm = [&](Frag& a, int kt) {
    Frag b[4];
#pragma unroll
    for (int j = 0; j < 4; ++j) {
      const u32x4* bq = (const u32x4*)(bbase + ((size_t)kt * 16 + j) * 32 * 8);
      b[j].q[0] = bq[0];
      b[j].q[1] = bq[1];
    }
#pragma unroll
    for (int j = 0; j < 4; ++j)
      acc[j] = __builtin_amdgcn_wmma_f32_16x16x32_bf16(
          false, a.v, false, b[j].v, (short)0, acc[j], false, false);
  };

  const int nk = K >> 5;  // 4 or 8 (always even)
  loadA(a0, 0);
  for (int kt = 0; kt + 2 <= nk; kt += 2) {
    loadA(a1, kt + 1);
    mm(a0, kt);
    if (kt + 2 < nk) loadA(a0, kt + 2);
    mm(a1, kt + 1);
  }

  const int rbase = tm * 16 + hi * 8;
#pragma unroll
  for (int j = 0; j < 4; ++j) {
    const int col = (wave * 4 + j) * 16 + (lane & 15);
#pragma unroll
    for (int r = 0; r < 8; ++r)
      C[(size_t)(rbase + r) * DMODEL + col] = acc[j][r];
  }
}

// ---------------- attention logits: al/ar [N,H], one wave per (node,head) ----------------
__global__ __launch_bounds__(256) void attn_logits_k(
    const float* __restrict__ h, const float* __restrict__ asrc,
    const float* __restrict__ adst, float* __restrict__ al, float* __restrict__ ar,
    int H, int C) {
  int gw = (blockIdx.x * blockDim.x + threadIdx.x) >> 5;
  int lane = threadIdx.x & 31;
  if (gw >= NNODES * H) return;
  int hh = gw % H;
  int n  = gw / H;
  const float* hp = h + (size_t)n * DMODEL + hh * C;
  const float* ap = asrc + hh * C;
  const float* bp = adst + hh * C;
  float sa = 0.f, sb = 0.f;
  for (int c = lane; c < C; c += 32) {
    float v = hp[c];
    sa += v * ap[c];
    sb += v * bp[c];
  }
  for (int o = 16; o; o >>= 1) {
    sa += __shfl_xor(sa, o, 32);
    sb += __shfl_xor(sb, o, 32);
  }
  if (lane == 0) { al[gw] = sa; ar[gw] = sb; }
}

// ---------------- edge pass 1: leaky-relu logit + per-dst max (ordered-uint atomicMax) ----
__global__ void edge_max_k(const int* __restrict__ ei, const float* __restrict__ al,
                           const float* __restrict__ ar, float* __restrict__ ebuf,
                           unsigned* __restrict__ mu, int H) {
  int t = blockIdx.x * blockDim.x + threadIdx.x;
  if (t >= ETOT * H) return;
  int e = t / H, hh = t - e * H;
  int s, d;
  if (e < NEDGES) { s = ei[e]; d = ei[NEDGES + e]; } else { s = d = e - NEDGES; }
  float v = al[s * H + hh] + ar[d * H + hh];
  v = v > 0.f ? v : SLOPE * v;
  ebuf[t] = v;
  atomicMax(&mu[d * H + hh], ordf(v));
}

// ---------------- edge pass 2: p = exp(e - m[dst]); s[dst] += p ----------------
__global__ void edge_exp_k(const int* __restrict__ ei, float* __restrict__ ebuf,
                           const unsigned* __restrict__ mu, float* __restrict__ ssum, int H) {
  int t = blockIdx.x * blockDim.x + threadIdx.x;
  if (t >= ETOT * H) return;
  int e = t / H, hh = t - e * H;
  int d = (e < NEDGES) ? ei[NEDGES + e] : (e - NEDGES);
  float m = deordf(mu[d * H + hh]);
  float p = __expf(ebuf[t] - m);
  ebuf[t] = p;
  atomicAdd(&ssum[d * H + hh], p);
}

// ---------------- edge pass 3: out[dst] += alpha * h[src]; one wave per edge ----------------
__global__ __launch_bounds__(256) void edge_aggregate_k(
    const int* __restrict__ ei, const float* __restrict__ ebuf,
    const float* __restrict__ ssum, const float* __restrict__ h,
    float* __restrict__ out, int H, int Csh) {
  int gw = (blockIdx.x * blockDim.x + threadIdx.x) >> 5;
  int lane = threadIdx.x & 31;
  if (gw >= ETOT) return;
  int e = gw;
  int s, d;
  if (e < NEDGES) { s = ei[e]; d = ei[NEDGES + e]; } else { s = d = e - NEDGES; }
  const float* hs = h + (size_t)s * DMODEL;
  float* od = out + (size_t)d * DMODEL;
#pragma unroll
  for (int j = 0; j < 8; ++j) {
    int f = lane + 32 * j;
    int hh = f >> Csh;
    float alpha = ebuf[(size_t)e * H + hh] / (ssum[(size_t)d * H + hh] + 1e-16f);
    atomicAdd(&od[f], alpha * hs[f]);
  }
}

// ---------------- BN: coalesced partial column sums ----------------
__global__ __launch_bounds__(256) void bn_partial_k(const float* __restrict__ x,
                                                    const float* __restrict__ bias,
                                                    float* __restrict__ sum,
                                                    float* __restrict__ sq) {
  int col = threadIdx.x;
  float b = bias[col];
  float s = 0.f, s2 = 0.f;
  for (int r = blockIdx.x; r < NNODES; r += gridDim.x) {
    float v = x[(size_t)r * DMODEL + col] + b;
    s += v;
    s2 += v * v;
  }
  atomicAdd(&sum[col], s);
  atomicAdd(&sq[col], s2);
}

__global__ void bn_finalize_k(const float* __restrict__ sum, const float* __restrict__ sq,
                              const float* __restrict__ g, const float* __restrict__ be,
                              float* __restrict__ bna, float* __restrict__ bnb) {
  int col = threadIdx.x;
  float mu  = sum[col] / (float)NNODES;
  float var = sq[col] / (float)NNODES - mu * mu;
  float a = rsqrtf(var + 1e-5f) * g[col];
  bna[col] = a;
  bnb[col] = be[col] - mu * a;
}

// ---------------- BN apply fused with bf16 recast for the next GEMM ----------------
__global__ void bn_apply_k(const float* __restrict__ x, const float* __restrict__ bias,
                           const float* __restrict__ a, const float* __restrict__ b,
                           float* __restrict__ y, bf16* __restrict__ ybf) {
  size_t i = (size_t)blockIdx.x * blockDim.x + threadIdx.x;
  if (i >= (size_t)NNODES * DMODEL) return;
  int col = (int)(i & (DMODEL - 1));
  float v = (x[i] + bias[col]) * a[col] + b[col];
  y[i] = v;
  ybf[i] = (bf16)v;
}

// ---------------- final linear (256->2) + 2-class softmax; one wave per node ----------------
__global__ __launch_bounds__(256) void final_head_k(const float* __restrict__ h,
                                                    const float* __restrict__ lnW,
                                                    const float* __restrict__ lnb,
                                                    float* __restrict__ out) {
  int gw = (blockIdx.x * blockDim.x + threadIdx.x) >> 5;
  int lane = threadIdx.x & 31;
  if (gw >= NNODES) return;
  const float* hp = h + (size_t)gw * DMODEL;
  float d0 = 0.f, d1 = 0.f;
  for (int k = lane; k < DMODEL; k += 32) {
    float v = hp[k];
    d0 += v * lnW[k * 2];
    d1 += v * lnW[k * 2 + 1];
  }
  for (int o = 16; o; o >>= 1) {
    d0 += __shfl_xor(d0, o, 32);
    d1 += __shfl_xor(d1, o, 32);
  }
  if (lane == 0) {
    d0 += lnb[0];
    d1 += lnb[1];
    float m  = fmaxf(d0, d1);
    float p0 = __expf(d0 - m), p1 = __expf(d1 - m);
    float inv = 1.f / (p0 + p1);
    out[(size_t)gw * 2]     = p0 * inv;
    out[(size_t)gw * 2 + 1] = p1 * inv;
  }
}

extern "C" void kernel_launch(void* const* d_in, const int* in_sizes, int n_in,
                              void* d_out, int out_size, void* d_ws, size_t ws_size,
                              hipStream_t stream) {
  (void)in_sizes; (void)n_in; (void)out_size; (void)ws_size;
  const float* x  = (const float*)d_in[0];
  const int*   ei = (const int*)d_in[1];
  const float* W[3]  = {(const float*)d_in[2],  (const float*)d_in[6],  (const float*)d_in[10]};
  const float* AS[3] = {(const float*)d_in[3],  (const float*)d_in[7],  (const float*)d_in[11]};
  const float* AD[3] = {(const float*)d_in[4],  (const float*)d_in[8],  (const float*)d_in[12]};
  const float* BI[3] = {(const float*)d_in[5],  (const float*)d_in[9],  (const float*)d_in[13]};
  const float* G[3]  = {(const float*)d_in[14], (const float*)d_in[16], (const float*)d_in[18]};
  const float* BE[3] = {(const float*)d_in[15], (const float*)d_in[17], (const float*)d_in[19]};
  const float* lnW = (const float*)d_in[20];
  const float* lnb = (const float*)d_in[21];
  float* out = (float*)d_out;

  char* p = (char*)d_ws;
  auto alloc = [&](size_t bytes) -> char* {
    char* r = p;
    p += (bytes + 255) & ~(size_t)255;
    return r;
  };
  bf16*     Abf  = (bf16*)alloc((size_t)NNODES * DMODEL * 2);
  float*    hW   = (float*)alloc((size_t)NNODES * DMODEL * 4);
  float*    agg  = (float*)alloc((size_t)NNODES * DMODEL * 4);
  float*    ebuf = (float*)alloc((size_t)ETOT * 4 * 4);
  float*    al   = (float*)alloc((size_t)NNODES * 4 * 4);
  float*    ar   = (float*)alloc((size_t)NNODES * 4 * 4);
  unsigned* mu   = (unsigned*)alloc((size_t)NNODES * 4 * 4);
  float*    ssum = (float*)alloc((size_t)NNODES * 4 * 4);
  uint32_t* Bp   = (uint32_t*)alloc((size_t)DMODEL * DMODEL * 2);
  float*    bns  = (float*)alloc(DMODEL * 4);
  float*    bnq  = (float*)alloc(DMODEL * 4);
  float*    bna  = (float*)alloc(DMODEL * 4);
  float*    bnb  = (float*)alloc(DMODEL * 4);

  // layer 0 input cast: x (N x 128) -> bf16
  {
    int n = NNODES * 128;
    cast_bf16_k<<<(n + 255) / 256, 256, 0, stream>>>(x, Abf, n);
  }

  const int Ks[3]  = {128, DMODEL, DMODEL};
  const int Hs[3]  = {1, 4, 4};
  const int Csh[3] = {8, 6, 6};  // log2(C), C = 256/H

  for (int l = 0; l < 3; ++l) {
    const int K = Ks[l], H = Hs[l], C = DMODEL / H;

    { int n = (K >> 5) * 16 * 32 * 8;
      pack_B_k<<<(n + 255) / 256, 256, 0, stream>>>(W[l], Bp, K); }

    { size_t ldsBytes = (size_t)K * 512;  // packed B for whole K loop
      gemm_bf16_wmma<<<NNODES / 16, 128, ldsBytes, stream>>>(Abf, Bp, hW, K); }

    { int waves = NNODES * H;
      attn_logits_k<<<(waves * 32 + 255) / 256, 256, 0, stream>>>(hW, AS[l], AD[l], al, ar, H, C); }

    (void)hipMemsetAsync(mu,   0, (size_t)NNODES * H * 4, stream);
    (void)hipMemsetAsync(ssum, 0, (size_t)NNODES * H * 4, stream);
    (void)hipMemsetAsync(agg,  0, (size_t)NNODES * DMODEL * 4, stream);

    { int t = ETOT * H;
      edge_max_k<<<(t + 255) / 256, 256, 0, stream>>>(ei, al, ar, ebuf, mu, H);
      edge_exp_k<<<(t + 255) / 256, 256, 0, stream>>>(ei, ebuf, mu, ssum, H); }

    { long long thr = (long long)ETOT * 32;
      edge_aggregate_k<<<(int)((thr + 255) / 256), 256, 0, stream>>>(ei, ebuf, ssum, hW, agg, H, Csh[l]); }

    (void)hipMemsetAsync(bns, 0, DMODEL * 4, stream);
    (void)hipMemsetAsync(bnq, 0, DMODEL * 4, stream);
    bn_partial_k<<<512, 256, 0, stream>>>(agg, BI[l], bns, bnq);
    bn_finalize_k<<<1, 256, 0, stream>>>(bns, bnq, G[l], BE[l], bna, bnb);

    { size_t n = (size_t)NNODES * DMODEL;
      bn_apply_k<<<(int)((n + 255) / 256), 256, 0, stream>>>(agg, BI[l], bna, bnb, hW, Abf); }
  }

  { long long thr = (long long)NNODES * 32;
    final_head_k<<<(int)((thr + 255) / 256), 256, 0, stream>>>(hW, lnW, lnb, out); }
}
